// SparseMoEExpert_68667937129032
// MI455X (gfx1250) — matrix operand order repaired and checked by reading, not compiled, not run on hardware
//
#include <hip/hip_runtime.h>
#include <cstdint>
#include <cstddef>

#define N_TOK 32768
#define DIM   1024
#define HDIM  512
#define CDIM  512
#define NEXP  8
#define TOPK  2
#define NK    (N_TOK * TOPK)
#define KCH   64     // K-chunk per pipeline stage (2 WMMA sub-steps of 32)
#define LDSTR 72     // padded LDS row stride (elements): 144B, 16B-aligned, conflict-free

#if defined(__has_builtin)
#  if __has_builtin(__builtin_amdgcn_global_load_async_to_lds_b128)
#    define HAVE_ASYNC_CP 1
#  endif
#  if __has_builtin(__builtin_amdgcn_s_wait_asynccnt)
#    define HAVE_WAIT_ASYNC_BUILTIN 1
#  endif
#endif

#if HAVE_ASYNC_CP
#  define NBUF 3
#else
#  define NBUF 2
#endif

typedef __attribute__((ext_vector_type(16))) __bf16       v16bf;
typedef __attribute__((ext_vector_type(8)))  float        v8f;
typedef __attribute__((ext_vector_type(8)))  unsigned int uint8v;
typedef __attribute__((ext_vector_type(2)))  float        f32x2;
typedef __attribute__((ext_vector_type(2)))  __bf16       bf16x2;
typedef int v4i __attribute__((vector_size(16)));

static __device__ __forceinline__ unsigned int cvt2(float a, float b) {
    f32x2 f = {a, b};
    bf16x2 h = __builtin_convertvector(f, bf16x2);   // native packed f32->bf16
    return __builtin_bit_cast(unsigned int, h);
}
static __device__ __forceinline__ unsigned short cvt1(float a) {
    __bf16 h = (__bf16)a;
    return __builtin_bit_cast(unsigned short, h);
}
static __device__ __forceinline__ v16bf frag_from(const unsigned short* p0,
                                                  const unsigned short* p1) {
    uint4 lo = *reinterpret_cast<const uint4*>(p0);
    uint4 hi = *reinterpret_cast<const uint4*>(p1);
    uint8v v = {lo.x, lo.y, lo.z, lo.w, hi.x, hi.y, hi.z, hi.w};
    return __builtin_bit_cast(v16bf, v);
}

#if HAVE_ASYNC_CP
static __device__ __forceinline__ void async_cp16(unsigned short* lds,
                                                  const unsigned short* g) {
    __builtin_amdgcn_global_load_async_to_lds_b128(
        (__attribute__((address_space(1))) v4i*)g,
        (__attribute__((address_space(3))) v4i*)lds, 0, 0);
}
static __device__ __forceinline__ void wait_async0() {
#if HAVE_WAIT_ASYNC_BUILTIN
    __builtin_amdgcn_s_wait_asynccnt(0);
#else
    asm volatile("s_wait_asynccnt 0x0" ::: "memory");
#endif
}
static __device__ __forceinline__ void wait_async8() {   // allow 1 stage in flight
#if HAVE_WAIT_ASYNC_BUILTIN
    __builtin_amdgcn_s_wait_asynccnt(8);
#else
    asm volatile("s_wait_asynccnt 0x8" ::: "memory");
#endif
}
#endif

// 2 sub-steps of K=32 over one staged K=64 chunk; 16 WMMAs total
static __device__ __forceinline__ void compute_chunk(
    const unsigned short* __restrict__ lA, const unsigned short* __restrict__ lB,
    int wm, int wn, int laneN, int laneH, v8f acc[2][4]) {
#pragma unroll
    for (int ss = 0; ss < 2; ++ss) {
        v16bf af[2], bfr[4];
#pragma unroll
        for (int mt = 0; mt < 2; ++mt) {
            int row = wm * 32 + mt * 16 + laneN;
            int kO  = ss * 32 + (laneH ? 8 : 0);
            af[mt] = frag_from(&lA[row * LDSTR + kO], &lA[row * LDSTR + kO + 16]);
        }
#pragma unroll
        for (int nt = 0; nt < 4; ++nt) {
            int col = wn * 64 + nt * 16 + laneN;
            int kO  = ss * 32 + (laneH ? 8 : 0);
            bfr[nt] = frag_from(&lB[col * LDSTR + kO], &lB[col * LDSTR + kO + 16]);
        }
#pragma unroll
        for (int mt = 0; mt < 2; ++mt)
#pragma unroll
            for (int nt = 0; nt < 4; ++nt)
                acc[mt][nt] = __builtin_amdgcn_wmma_f32_16x16x32_bf16(
                    false, af[mt], false, bfr[nt], (short)0, acc[mt][nt], false, false);
    }
}

// ---------------------------------------------------------------- init ----
__global__ __launch_bounds__(256) void k_init(float* __restrict__ out,
                                              int* __restrict__ ctrl) {
    int idx = blockIdx.x * 256 + threadIdx.x;           // N*C/4 threads exactly
    float4 z = {0.f, 0.f, 0.f, 0.f};
    reinterpret_cast<float4*>(out)[idx] = z;
    if (blockIdx.x == 0 && threadIdx.x < 24) ctrl[threadIdx.x] = 0;
}

// ---------------------------------------------------------------- gate ----
// one wave32 per token; also converts x -> bf16 (rides on the bandwidth-bound pass)
__global__ __launch_bounds__(256) void k_gate(const float* __restrict__ x,
                                              const float* __restrict__ Wg,
                                              const float* __restrict__ bg,
                                              int* __restrict__ counts,
                                              int* __restrict__ top_idx,
                                              float* __restrict__ top_gate,
                                              unsigned short* __restrict__ xbf) {
    __shared__ float sWg[DIM * NEXP];
    int tid = threadIdx.x;
    for (int i = tid; i < DIM * NEXP; i += 256) sWg[i] = Wg[i];
    __syncthreads();

    int lane = tid & 31, wid = tid >> 5;
    int n = blockIdx.x * 8 + wid;
    const float4* x4 = reinterpret_cast<const float4*>(x + (size_t)n * DIM);
    unsigned short* xo = xbf + (size_t)n * DIM;

    float acc[NEXP];
#pragma unroll
    for (int e = 0; e < NEXP; ++e) acc[e] = 0.f;

    for (int i = 0; i < DIM / 128; ++i) {               // 8 iters, b128 loads
        float4 xv = x4[i * 32 + lane];
        uint2 u;
        u.x = cvt2(xv.x, xv.y);
        u.y = cvt2(xv.z, xv.w);
        *reinterpret_cast<uint2*>(&xo[(i * 32 + lane) * 4]) = u;
        float xa[4] = {xv.x, xv.y, xv.z, xv.w};
        int d = (i * 32 + lane) * 4;
#pragma unroll
        for (int j = 0; j < 4; ++j) {
            const float* wr = &sWg[(d + j) * NEXP];
#pragma unroll
            for (int e = 0; e < NEXP; ++e) acc[e] += xa[j] * wr[e];
        }
    }
#pragma unroll
    for (int off = 16; off >= 1; off >>= 1) {
#pragma unroll
        for (int e = 0; e < NEXP; ++e) acc[e] += __shfl_xor(acc[e], off, 32);
    }
    if (lane == 0) {
        float v0 = -3.4e38f, v1 = -3.4e38f; int i0 = 0, i1 = 0;
#pragma unroll
        for (int e = 0; e < NEXP; ++e) {
            float l = acc[e] + bg[e];
            if (l > v0)      { v1 = v0; i1 = i0; v0 = l; i0 = e; }
            else if (l > v1) { v1 = l;  i1 = e; }
        }
        float e1  = __expf(v1 - v0);
        float inv = 1.f / (1.f + e1);
        top_idx [2 * n + 0] = i0;  top_idx [2 * n + 1] = i1;
        top_gate[2 * n + 0] = inv; top_gate[2 * n + 1] = e1 * inv;
        atomicAdd(&counts[i0], 1);
        atomicAdd(&counts[i1], 1);
    }
}

// --------------------------------------------- weight convert+transpose ----
// src: [E][R][C] f32  ->  dst: [E][C][R] bf16   (64x64 tiles via LDS)
__global__ __launch_bounds__(256) void k_wcvt(const float* __restrict__ src,
                                              unsigned short* __restrict__ dst,
                                              int R, int C) {
    const int e = blockIdx.z;
    const int rBase = blockIdx.x * 64, cBase = blockIdx.y * 64;
    __shared__ float t[64][65];
    const float* s = src + (size_t)e * R * C;
    unsigned short* d = dst + (size_t)e * R * C;
    const int tid = threadIdx.x;
    const int r0 = tid >> 2, cs = tid & 3;

    const float4* g = reinterpret_cast<const float4*>(
        s + (size_t)(rBase + r0) * C + cBase + cs * 16);
#pragma unroll
    for (int q = 0; q < 4; ++q) {
        float4 v = g[q];
        t[r0][cs * 16 + q * 4 + 0] = v.x;
        t[r0][cs * 16 + q * 4 + 1] = v.y;
        t[r0][cs * 16 + q * 4 + 2] = v.z;
        t[r0][cs * 16 + q * 4 + 3] = v.w;
    }
    __syncthreads();
    // output row = c-index (cBase + r0), cols = r-indices (rBase + cs*16 ..)
    unsigned short* o = d + (size_t)(cBase + r0) * R + rBase + cs * 16;
#pragma unroll
    for (int q = 0; q < 2; ++q) {
        uint4 u;
        u.x = cvt2(t[cs * 16 + q * 8 + 0][r0], t[cs * 16 + q * 8 + 1][r0]);
        u.y = cvt2(t[cs * 16 + q * 8 + 2][r0], t[cs * 16 + q * 8 + 3][r0]);
        u.z = cvt2(t[cs * 16 + q * 8 + 4][r0], t[cs * 16 + q * 8 + 5][r0]);
        u.w = cvt2(t[cs * 16 + q * 8 + 6][r0], t[cs * 16 + q * 8 + 7][r0]);
        reinterpret_cast<uint4*>(o)[q] = u;
    }
}

// -------------------------------------------------------------- prefix ----
__global__ void k_prefix(const int* __restrict__ counts,
                         int* __restrict__ offsets, int* __restrict__ cursor) {
    if (threadIdx.x == 0) {
        int off = 0;
        for (int e = 0; e < NEXP; ++e) {
            offsets[e] = off; cursor[e] = off; off += counts[e];
        }
    }
}

// ------------------------------------------------------------- scatter ----
__global__ __launch_bounds__(256) void k_scatter(const int* __restrict__ top_idx,
                                                 const float* __restrict__ top_gate,
                                                 int* __restrict__ cursor,
                                                 int* __restrict__ slot_token,
                                                 float* __restrict__ slot_gate) {
    int n = blockIdx.x * 256 + threadIdx.x;
#pragma unroll
    for (int k = 0; k < TOPK; ++k) {
        int e = top_idx[2 * n + k];
        int pos = atomicAdd(&cursor[e], 1);
        slot_token[pos] = n;
        slot_gate[pos]  = top_gate[2 * n + k];
    }
}

// ------------------------------------------------------ FFN layer 1 -------
// A: gathered bf16 x rows; B: pre-transposed bf16 W1T[e][h][d].
// grid: (4*(N/128), E); bx = tile*4 + z (z fastest => A rows L2-resident)
__global__ __launch_bounds__(256)
__attribute__((amdgpu_waves_per_eu(1)))
void k_ffn1(const unsigned short* __restrict__ xbf,
            const unsigned short* __restrict__ W1T,
            const float* __restrict__ b1,
            const int* __restrict__ offsets,
            const int* __restrict__ counts,
            const int* __restrict__ slot_token,
            unsigned short* __restrict__ Hbuf) {
    const int e    = blockIdx.y;
    const int tile = blockIdx.x >> 2;
    const int hBase = (blockIdx.x & 3) * 128;
    const int off = offsets[e], cnt = counts[e];
    const int start = off + tile * 128;
    if (start >= off + cnt) return;
    const int valid = min(128, off + cnt - start);

    __shared__ unsigned short ldsA[NBUF][128 * LDSTR];
    __shared__ unsigned short ldsB[NBUF][128 * LDSTR];

    const int tid = threadIdx.x, lane = tid & 31, wid = tid >> 5;
    const int wm = wid & 3, wn = wid >> 2;
    const int laneN = lane & 15, laneH = lane >> 4;

    // staging roles (identical for A and B): row/col = tid>>1, half = tid&1
    const int rr = tid >> 1, half = tid & 1;
    const int tokA = (rr < valid) ? slot_token[start + rr] : slot_token[start];
    const unsigned short* pA = xbf + (size_t)tokA * DIM + half * 32;
    const unsigned short* pB = W1T + (size_t)e * DIM * HDIM
                                   + (size_t)(hBase + rr) * DIM + half * 32;

    v8f acc[2][4] = {};
    const int NST = DIM / KCH;

#if HAVE_ASYNC_CP
    auto stage = [&](int kc, unsigned short* bufA, unsigned short* bufB) {
        unsigned short* da = &bufA[rr * LDSTR + half * 32];
        unsigned short* db = &bufB[rr * LDSTR + half * 32];
#pragma unroll
        for (int q = 0; q < 4; ++q) {
            async_cp16(da + q * 8, pA + kc + q * 8);
            async_cp16(db + q * 8, pB + kc + q * 8);
        }
    };
    stage(0, ldsA[0], ldsB[0]);
    stage(KCH, ldsA[1], ldsB[1]);
    wait_async8();                      // chunk0 landed; chunk1 in flight
    __syncthreads();
    int cur = 0;
#pragma unroll 1
    for (int s = 0; s < NST; ++s) {
        int pre = cur + 2; if (pre >= 3) pre -= 3;
        if (s + 2 < NST) stage((s + 2) * KCH, ldsA[pre], ldsB[pre]);
        compute_chunk(ldsA[cur], ldsB[cur], wm, wn, laneN, laneH, acc);
        if (s + 2 < NST) wait_async8(); else wait_async0();
        __syncthreads();
        cur = (cur + 1 == 3) ? 0 : cur + 1;
    }
#else
    uint4 aU[4], bU[4];
    auto loadAB = [&](int kc) {
        const uint4* ga = reinterpret_cast<const uint4*>(pA + kc);
        const uint4* gb = reinterpret_cast<const uint4*>(pB + kc);
#pragma unroll
        for (int i = 0; i < 4; ++i) { aU[i] = ga[i]; bU[i] = gb[i]; }
    };
    auto storeAB = [&](unsigned short* bufA, unsigned short* bufB) {
        uint4* da = reinterpret_cast<uint4*>(&bufA[rr * LDSTR + half * 32]);
        uint4* db = reinterpret_cast<uint4*>(&bufB[rr * LDSTR + half * 32]);
#pragma unroll
        for (int q = 0; q < 4; ++q) { da[q] = aU[q]; db[q] = bU[q]; }
    };
    loadAB(0);
    storeAB(ldsA[0], ldsB[0]);
    __syncthreads();
#pragma unroll 1
    for (int s = 0; s < NST; ++s) {
        int cur = s & 1;
        if (s + 1 < NST) loadAB((s + 1) * KCH);
        compute_chunk(ldsA[cur], ldsB[cur], wm, wn, laneN, laneH, acc);
        if (s + 1 < NST) storeAB(ldsA[cur ^ 1], ldsB[cur ^ 1]);
        __syncthreads();
    }
#endif

    const float* b1e = b1 + e * HDIM;
    if (valid == 128) {
#pragma unroll
        for (int nt = 0; nt < 4; ++nt) {
            int col = hBase + wn * 64 + nt * 16 + laneN;
            float bias = b1e[col];
#pragma unroll
            for (int mt = 0; mt < 2; ++mt) {
                int rowBase = wm * 32 + mt * 16 + laneH * 8;
#pragma unroll
                for (int r = 0; r < 8; ++r) {
                    float v = acc[mt][nt][r] + bias;
                    v = v > 0.f ? v : 0.f;
                    Hbuf[(size_t)(start + rowBase + r) * HDIM + col] = cvt1(v);
                }
            }
        }
    } else {
#pragma unroll
        for (int nt = 0; nt < 4; ++nt) {
            int col = hBase + wn * 64 + nt * 16 + laneN;
            float bias = b1e[col];
#pragma unroll
            for (int mt = 0; mt < 2; ++mt) {
                int rowBase = wm * 32 + mt * 16 + laneH * 8;
#pragma unroll
                for (int r = 0; r < 8; ++r) {
                    int row = rowBase + r;
                    if (row < valid) {
                        float v = acc[mt][nt][r] + bias;
                        v = v > 0.f ? v : 0.f;
                        Hbuf[(size_t)(start + row) * HDIM + col] = cvt1(v);
                    }
                }
            }
        }
    }
}

// ------------------------------------------------------ FFN layer 2 -------
// A: bf16 H rows (contiguous slots); B: pre-transposed bf16 W2T[e][c][h].
// out += gate * (H @ W2 + b2); exactly 2 commutative f32 atomics per element.
__global__ __launch_bounds__(256)
__attribute__((amdgpu_waves_per_eu(1)))
void k_ffn2(const unsigned short* __restrict__ Hbuf,
            const unsigned short* __restrict__ W2T,
            const float* __restrict__ b2,
            const int* __restrict__ offsets,
            const int* __restrict__ counts,
            const int* __restrict__ slot_token,
            const float* __restrict__ slot_gate,
            float* __restrict__ out) {
    const int e    = blockIdx.y;
    const int tile = blockIdx.x >> 2;
    const int cBase = (blockIdx.x & 3) * 128;
    const int off = offsets[e], cnt = counts[e];
    const int start = off + tile * 128;
    if (start >= off + cnt) return;
    const int valid = min(128, off + cnt - start);

    __shared__ unsigned short ldsA[NBUF][128 * LDSTR];
    __shared__ unsigned short ldsB[NBUF][128 * LDSTR];

    const int tid = threadIdx.x, lane = tid & 31, wid = tid >> 5;
    const int wm = wid & 3, wn = wid >> 2;
    const int laneN = lane & 15, laneH = lane >> 4;

    const int rr = tid >> 1, half = tid & 1;
    int slotA = start + rr; if (slotA >= NK) slotA = NK - 1;
    const unsigned short* pA = Hbuf + (size_t)slotA * HDIM + half * 32;
    const unsigned short* pB = W2T + (size_t)e * HDIM * CDIM
                                   + (size_t)(cBase + rr) * HDIM + half * 32;

    v8f acc[2][4] = {};
    const int NST = HDIM / KCH;

#if HAVE_ASYNC_CP
    auto stage = [&](int kc, unsigned short* bufA, unsigned short* bufB) {
        unsigned short* da = &bufA[rr * LDSTR + half * 32];
        unsigned short* db = &bufB[rr * LDSTR + half * 32];
#pragma unroll
        for (int q = 0; q < 4; ++q) {
            async_cp16(da + q * 8, pA + kc + q * 8);
            async_cp16(db + q * 8, pB + kc + q * 8);
        }
    };
    stage(0, ldsA[0], ldsB[0]);
    stage(KCH, ldsA[1], ldsB[1]);
    wait_async8();                      // chunk0 landed; chunk1 in flight
    __syncthreads();
    int cur = 0;
#pragma unroll 1
    for (int s = 0; s < NST; ++s) {
        int pre = cur + 2; if (pre >= 3) pre -= 3;
        if (s + 2 < NST) stage((s + 2) * KCH, ldsA[pre], ldsB[pre]);
        compute_chunk(ldsA[cur], ldsB[cur], wm, wn, laneN, laneH, acc);
        if (s + 2 < NST) wait_async8(); else wait_async0();
        __syncthreads();
        cur = (cur + 1 == 3) ? 0 : cur + 1;
    }
#else
    uint4 aU[4], bU[4];
    auto loadAB = [&](int kc) {
        const uint4* ga = reinterpret_cast<const uint4*>(pA + kc);
        const uint4* gb = reinterpret_cast<const uint4*>(pB + kc);
#pragma unroll
        for (int i = 0; i < 4; ++i) { aU[i] = ga[i]; bU[i] = gb[i]; }
    };
    auto storeAB = [&](unsigned short* bufA, unsigned short* bufB) {
        uint4* da = reinterpret_cast<uint4*>(&bufA[rr * LDSTR + half * 32]);
        uint4* db = reinterpret_cast<uint4*>(&bufB[rr * LDSTR + half * 32]);
#pragma unroll
        for (int q = 0; q < 4; ++q) { da[q] = aU[q]; db[q] = bU[q]; }
    };
    loadAB(0);
    storeAB(ldsA[0], ldsB[0]);
    __syncthreads();
#pragma unroll 1
    for (int s = 0; s < NST; ++s) {
        int cur = s & 1;
        if (s + 1 < NST) loadAB((s + 1) * KCH);
        compute_chunk(ldsA[cur], ldsB[cur], wm, wn, laneN, laneH, acc);
        if (s + 1 < NST) storeAB(ldsA[cur ^ 1], ldsB[cur ^ 1]);
        __syncthreads();
    }
#endif

    const float* b2e = b2 + e * CDIM;
    int   tok[2][8];
    float gw [2][8];
#pragma unroll
    for (int mt = 0; mt < 2; ++mt) {
        int rowBase = wm * 32 + mt * 16 + laneH * 8;
#pragma unroll
        for (int r = 0; r < 8; ++r) {
            int row = rowBase + r;
            if (row < valid) {
                int s = start + row;
                tok[mt][r] = slot_token[s];
                gw [mt][r] = slot_gate[s];
            } else { tok[mt][r] = 0; gw[mt][r] = 0.f; }
        }
    }
    if (valid == 128) {
#pragma unroll
        for (int nt = 0; nt < 4; ++nt) {
            int col = cBase + wn * 64 + nt * 16 + laneN;
            float bias = b2e[col];
#pragma unroll
            for (int mt = 0; mt < 2; ++mt)
#pragma unroll
                for (int r = 0; r < 8; ++r)
                    atomicAdd(&out[(size_t)tok[mt][r] * CDIM + col],
                              gw[mt][r] * (acc[mt][nt][r] + bias));
        }
    } else {
#pragma unroll
        for (int nt = 0; nt < 4; ++nt) {
            int col = cBase + wn * 64 + nt * 16 + laneN;
            float bias = b2e[col];
#pragma unroll
            for (int mt = 0; mt < 2; ++mt) {
                int rowBase = wm * 32 + mt * 16 + laneH * 8;
#pragma unroll
                for (int r = 0; r < 8; ++r) {
                    if (rowBase + r < valid)
                        atomicAdd(&out[(size_t)tok[mt][r] * CDIM + col],
                                  gw[mt][r] * (acc[mt][nt][r] + bias));
                }
            }
        }
    }
}

// ---------------------------------------------------------------- host ----
extern "C" void kernel_launch(void* const* d_in, const int* in_sizes, int n_in,
                              void* d_out, int out_size, void* d_ws, size_t ws_size,
                              hipStream_t stream) {
    const float* x  = (const float*)d_in[0];
    const float* W1 = (const float*)d_in[1];
    const float* b1 = (const float*)d_in[2];
    const float* W2 = (const float*)d_in[3];
    const float* b2 = (const float*)d_in[4];
    const float* Wg = (const float*)d_in[5];
    const float* bg = (const float*)d_in[6];
    float* out = (float*)d_out;

    const size_t MB = 1u << 20;
    uint8_t* w = (uint8_t*)d_ws;
    int*   counts     = (int*)(w + 0);
    int*   offsets    = (int*)(w + 32);
    int*   cursor     = (int*)(w + 64);
    int*   top_idx    = (int*)(w + 128);
    float* top_gate   = (float*)(w + 128 + 4 * (size_t)NK);
    int*   slot_token = (int*)(w + 128 + 8 * (size_t)NK);
    float* slot_gate  = (float*)(w + 128 + 12 * (size_t)NK);
    unsigned short* xbf  = (unsigned short*)(w + 2 * MB);     //  64 MB
    unsigned short* Hbuf = (unsigned short*)(w + 68 * MB);    //  64 MB
    unsigned short* W1T  = (unsigned short*)(w + 133 * MB);   // 8.4 MB
    unsigned short* W2T  = (unsigned short*)(w + 142 * MB);   // 4.2 MB

    k_init   <<<dim3((N_TOK * CDIM) / 1024), 256, 0, stream>>>(out, counts);
    k_gate   <<<dim3(N_TOK / 8),  256, 0, stream>>>(x, Wg, bg, counts, top_idx,
                                                    top_gate, xbf);
    k_wcvt   <<<dim3(DIM / 64, HDIM / 64, NEXP), 256, 0, stream>>>(W1, W1T, DIM, HDIM);
    k_wcvt   <<<dim3(HDIM / 64, CDIM / 64, NEXP), 256, 0, stream>>>(W2, W2T, HDIM, CDIM);
    k_prefix <<<dim3(1),           32, 0, stream>>>(counts, offsets, cursor);
    k_scatter<<<dim3(N_TOK / 256), 256, 0, stream>>>(top_idx, top_gate, cursor,
                                                     slot_token, slot_gate);
    k_ffn1   <<<dim3(4 * (N_TOK / 128), NEXP), 256, 0, stream>>>(
        xbf, W1T, b1, offsets, counts, slot_token, Hbuf);
    k_ffn2   <<<dim3(4 * (N_TOK / 128), NEXP), 256, 0, stream>>>(
        Hbuf, W2T, b2, offsets, counts, slot_token, slot_gate, out);
}
